// AppearanceLoss_47296179863822
// MI455X (gfx1250) — matrix operand order repaired and checked by reading, compiled, not run on hardware
//
#include <hip/hip_runtime.h>
#include <hip/hip_bf16.h>

typedef __attribute__((ext_vector_type(16))) _Float16 v16h;
typedef __attribute__((ext_vector_type(8)))  float    v8f;
typedef __attribute__((ext_vector_type(4)))  unsigned int v4u;

union FragU { v16h h; v4u u[2]; };

// ---- async staging helper: copy tile rows x 32 halves (row-contiguous in K) ----
// Fast path: global_load_async_to_lds_b128 per 16B granule (ASYNCcnt-tracked).
// Fallback (ragged edges): guarded scalar stores incl. zero-fill.
__device__ __forceinline__ void stage_tile_async(
    const _Float16* __restrict__ src, _Float16 (*dst)[40],
    int rows, int r0, int Rmax, long ldk, int k0, int K, int tid, int nthr) {
  int granules = rows * 4;  // 4 x 16B per 32-half row
  for (int e = tid; e < granules; e += nthr) {
    int rr = e >> 2, kk = (e & 3) * 8;
    int row = r0 + rr;
    if (row < Rmax && k0 + kk + 8 <= K) {
      const _Float16* gp = src + (size_t)row * ldk + k0 + kk;
      unsigned lds = (unsigned)(unsigned long long)&dst[rr][kk];
      asm volatile("global_load_async_to_lds_b128 %0, %1, off"
                   :: "v"(lds), "v"(gp) : "memory");
    } else {
#pragma unroll
      for (int t = 0; t < 8; ++t) {
        int k = k0 + kk + t;
        dst[rr][kk + t] = (row < Rmax && k < K)
                              ? src[(size_t)row * ldk + k] : (_Float16)0.f;
      }
    }
  }
}

__device__ __forceinline__ void wait_async_lds() {
  asm volatile("s_wait_asynccnt 0x0" ::: "memory");
}

// ---------------- image prep: normalize + f32->f16 (9 images, NCHW) -------------
__global__ void k_prep(const float* __restrict__ gen, const float* __restrict__ tgt,
                       _Float16* __restrict__ act, int HW) {
  const float mean[3] = {0.485f, 0.456f, 0.406f};
  const float sd[3]   = {0.229f, 0.224f, 0.225f};
  long total = 9L * 3 * HW;
  for (long idx = (long)blockIdx.x * blockDim.x + threadIdx.x; idx < total;
       idx += (long)gridDim.x * blockDim.x) {
    int  img = (int)(idx / (3L * HW));
    long r   = idx - (long)img * 3 * HW;
    int  c   = (int)(r / HW);
    float v  = (img < 8) ? gen[(long)img * 3 * HW + r] : tgt[r];
    act[idx] = (_Float16)((v - mean[c]) / sd[c]);
  }
}

// ------- weight repack: OIHW f32 -> [cout][k=ci*9+kh*3+kw] f16 (row = K-contig) -
__global__ void k_packw(const float* __restrict__ w, _Float16* __restrict__ wp,
                        int Cin, int Cout) {
  int K = Cin * 9;
  long total = (long)Cout * K;
  for (long idx = (long)blockIdx.x * blockDim.x + threadIdx.x; idx < total;
       idx += (long)gridDim.x * blockDim.x) {
    int k  = (int)(idx % K);
    int o  = (int)(idx / K);
    int ci = k / 9, rem = k - ci * 9, r = rem / 3, cc = rem - r * 3;
    wp[idx] = (_Float16)w[(((long)o * Cin + ci) * 3 + r) * 3 + cc];
  }
}

// ---------------- conv3x3 pad=1 as implicit GEMM via WMMA f16 -------------------
// block: 256 thr = 8 waves; tile 128 pixels x 64 couts; wave = 32x32 C subtile.
__launch_bounds__(256)
__global__ void k_conv3x3(const _Float16* __restrict__ X, const _Float16* __restrict__ Wp,
                          const float* __restrict__ bias, _Float16* __restrict__ Y,
                          int Cin, int Cout, int H, int W) {
  const int HW = H * W;
  const int K  = Cin * 9;
  const int img = blockIdx.z;
  const int m0  = blockIdx.x * 128;
  const int n0  = blockIdx.y * 64;
  const int tid = threadIdx.x, lane = tid & 31, wave = tid >> 5;
  const int wm = (wave & 3) * 32, wn = (wave >> 2) * 32;
  __shared__ _Float16 As[128][40];
  __shared__ _Float16 Bs[64][40];
  v8f acc[2][2] = {};
  const _Float16* Ximg = X + (size_t)img * Cin * HW;

  for (int k0 = 0; k0 < K; k0 += 32) {
    // stage A (im2col gather, non-contiguous -> sync path) 128x32
    for (int e = tid; e < 128 * 32; e += 256) {
      int p = e >> 5, kk = e & 31;
      int k = k0 + kk, pix = m0 + p;
      _Float16 v = (_Float16)0.f;
      if (k < K && pix < HW) {
        int ci = k / 9, rem = k - ci * 9, r = rem / 3, cc = rem - r * 3;
        int y = pix / W, x = pix - y * W;
        int iy = y + r - 1, ix = x + cc - 1;
        if (iy >= 0 && iy < H && ix >= 0 && ix < W)
          v = Ximg[(size_t)ci * HW + (size_t)iy * W + ix];
      }
      As[p][kk] = v;
    }
    // stage B (packed weights, K-contiguous rows -> async-to-LDS) 64x32
    stage_tile_async(Wp, Bs, 64, n0, Cout, K, k0, K, tid, 256);
    wait_async_lds();
    __syncthreads();

    const int l15 = lane & 15;
    const int kbA = (lane < 16) ? 0 : 8;   // ISA 16-bit A 16x32 lane layout
    const int kbB = (lane < 16) ? 0 : 8;   // B: same K layout as A
#pragma unroll
    for (int i = 0; i < 2; i++) {
      FragU ua;
      int mr = wm + i * 16 + l15;
      ua.u[0] = *(const v4u*)&As[mr][kbA];
      ua.u[1] = *(const v4u*)&As[mr][kbA + 16];
#pragma unroll
      for (int j = 0; j < 2; j++) {
        FragU ub;
        int nr = wn + j * 16 + l15;
        ub.u[0] = *(const v4u*)&Bs[nr][kbB];
        ub.u[1] = *(const v4u*)&Bs[nr][kbB + 16];
        acc[i][j] = __builtin_amdgcn_wmma_f32_16x16x32_f16(
            false, ua.h, false, ub.h, (short)0, acc[i][j], false, false);
      }
    }
    __syncthreads();
  }

  const int l15 = lane & 15, mb = (lane >> 4) * 8;
#pragma unroll
  for (int i = 0; i < 2; i++)
#pragma unroll
    for (int j = 0; j < 2; j++) {
      int nco = n0 + wn + j * 16 + l15;
      if (nco >= Cout) continue;
      float bv = bias[nco];
#pragma unroll
      for (int r = 0; r < 8; r++) {
        int pix = m0 + wm + i * 16 + mb + r;
        if (pix < HW) {
          float v = acc[i][j][r] + bv;
          v = fmaxf(v, 0.f);  // fused ReLU
          Y[((size_t)img * Cout + nco) * HW + pix] = (_Float16)v;
        }
      }
    }
}

// ---------------- generic batched GEMM-NT: C[z] = A[z](MxK) * B[z](NxK)^T ------
__launch_bounds__(256)
__global__ void k_gemm_nt(const _Float16* __restrict__ A, const _Float16* __restrict__ B,
                          float* __restrict__ C, int M, int N, int K,
                          long sA, long sB, long sC) {
  const int z = blockIdx.z;
  const _Float16* Az = A + (size_t)z * sA;
  const _Float16* Bz = B + (size_t)z * sB;
  float* Cz = C + (size_t)z * sC;
  const int m0 = blockIdx.x * 128, n0 = blockIdx.y * 64;
  const int tid = threadIdx.x, lane = tid & 31, wave = tid >> 5;
  const int wm = (wave & 3) * 32, wn = (wave >> 2) * 32;
  __shared__ _Float16 As[128][40];
  __shared__ _Float16 Bs[64][40];
  v8f acc[2][2] = {};

  for (int k0 = 0; k0 < K; k0 += 32) {
    // both operands are K-contiguous: async-to-LDS staging
    stage_tile_async(Az, As, 128, m0, M, K, k0, K, tid, 256);
    stage_tile_async(Bz, Bs, 64, n0, N, K, k0, K, tid, 256);
    wait_async_lds();
    __syncthreads();

    const int l15 = lane & 15;
    const int kbA = (lane < 16) ? 0 : 8;
    const int kbB = (lane < 16) ? 0 : 8;
#pragma unroll
    for (int i = 0; i < 2; i++) {
      FragU ua;
      int mr = wm + i * 16 + l15;
      ua.u[0] = *(const v4u*)&As[mr][kbA];
      ua.u[1] = *(const v4u*)&As[mr][kbA + 16];
#pragma unroll
      for (int j = 0; j < 2; j++) {
        FragU ub;
        int nr = wn + j * 16 + l15;
        ub.u[0] = *(const v4u*)&Bs[nr][kbB];
        ub.u[1] = *(const v4u*)&Bs[nr][kbB + 16];
        acc[i][j] = __builtin_amdgcn_wmma_f32_16x16x32_f16(
            false, ua.h, false, ub.h, (short)0, acc[i][j], false, false);
      }
    }
    __syncthreads();
  }

  const int l15 = lane & 15, mb = (lane >> 4) * 8;
#pragma unroll
  for (int i = 0; i < 2; i++)
#pragma unroll
    for (int j = 0; j < 2; j++) {
      int n = n0 + wn + j * 16 + l15;
      if (n >= N) continue;
#pragma unroll
      for (int r = 0; r < 8; r++) {
        int m = m0 + wm + i * 16 + mb + r;
        if (m < M) Cz[(size_t)m * N + n] = acc[i][j][r];
      }
    }
}

// ---------------- 2x2 average pool, fp16 ---------------------------------------
__global__ void k_pool(const _Float16* __restrict__ X, _Float16* __restrict__ Y,
                       int C, int H, int W) {
  int Ho = H >> 1, Wo = W >> 1;
  long total = 9L * C * Ho * Wo;
  for (long idx = (long)blockIdx.x * blockDim.x + threadIdx.x; idx < total;
       idx += (long)gridDim.x * blockDim.x) {
    int wo = (int)(idx % Wo); long t = idx / Wo;
    int ho = (int)(t % Ho);   t /= Ho;
    int c  = (int)(t % C);    int img = (int)(t / C);
    const _Float16* p = X + (((size_t)img * C + c) * H + (size_t)ho * 2) * W + (size_t)wo * 2;
    float s = (float)p[0] + (float)p[1] + (float)p[W] + (float)p[W + 1];
    Y[idx] = (_Float16)(0.25f * s);
  }
}

// ---------------- subsample ns pixels: xs[9][ns][C], xsT[9][C][ns], norms -------
__global__ void k_subsample(const _Float16* __restrict__ act, int C, int HW, int ns,
                            _Float16* __restrict__ xs, _Float16* __restrict__ xsT,
                            float* __restrict__ norms) {
  int bi = blockIdx.y;
  int i  = blockIdx.x * blockDim.x + threadIdx.x;
  if (i >= ns) return;
  unsigned h = (unsigned)(bi * 2654435761u) ^ (unsigned)(i * 40503u + 0x9E3779B9u);
  h ^= h >> 13; h *= 0x85EBCA6Bu; h ^= h >> 16;
  int idx = (int)(h % (unsigned)HW);
  float ss = 0.f;
  for (int ch = 0; ch < C; ++ch) {
    float v = (float)act[((size_t)bi * C + ch) * HW + idx];
    xs[((size_t)bi * ns + i) * C + ch]  = (_Float16)v;
    xsT[((size_t)bi * C + ch) * ns + i] = (_Float16)v;
    ss += v * v;
  }
  norms[(size_t)bi * ns + i] = sqrtf(ss);
}

// ---------------- per-channel mean over samples ---------------------------------
__global__ void k_mean(const _Float16* __restrict__ xsT, float* __restrict__ mu,
                       int C, int ns) {
  int bi = blockIdx.y;
  int ch = blockIdx.x * blockDim.x + threadIdx.x;
  if (ch >= C) return;
  const _Float16* p = xsT + ((size_t)bi * C + ch) * ns;
  float s = 0.f;
  for (int i = 0; i < ns; ++i) s += (float)p[i];
  mu[(size_t)bi * C + ch] = s / (float)ns;
}

// ---------------- center xsT in place -------------------------------------------
__global__ void k_center(_Float16* __restrict__ xsT, const float* __restrict__ mu,
                         int C, int ns) {
  long total = 9L * C * ns;
  for (long idx = (long)blockIdx.x * blockDim.x + threadIdx.x; idx < total;
       idx += (long)gridDim.x * blockDim.x) {
    long bc = idx / ns;
    xsT[idx] = (_Float16)((float)xsT[idx] - mu[bc]);
  }
}

// ---------------- cosine-distance REMD half: min over one axis ------------------
__global__ void k_cos_min(const float* __restrict__ dots, const float* __restrict__ norms,
                          float* __restrict__ acc, int ns, int mode) {
  int b = blockIdx.y;
  int i = blockIdx.x * blockDim.x + threadIdx.x;
  if (i >= ns) return;
  const float* yn = norms + (size_t)8 * ns;
  float best = 1e30f;
  if (mode == 0) {
    float nx = norms[(size_t)b * ns + i];
    const float* row = dots + ((size_t)b * ns + i) * ns;
    for (int m = 0; m < ns; ++m)
      best = fminf(best, 1.0f - row[m] / (nx * yn[m] + 1e-10f));
  } else {
    float ny = yn[i];
    for (int n = 0; n < ns; ++n) {
      float d = 1.0f - dots[((size_t)b * ns + n) * ns + i] /
                           (norms[(size_t)b * ns + n] * ny + 1e-10f);
      best = fminf(best, d);
    }
  }
  atomicAdd(&acc[b], best / (float)ns);
}

// ---------------- remd-combine + mean-diff --------------------------------------
__global__ void k_remd_mu(const float* __restrict__ accR, const float* __restrict__ accC,
                          const float* __restrict__ mu, float* __restrict__ loss, int C) {
  int b = threadIdx.x;
  if (b >= 8) return;
  float v = fmaxf(accR[b], accC[b]);
  if (C > 0) {
    const float* mx = mu + (size_t)b * C;
    const float* my = mu + (size_t)8 * C;
    float s = 0.f;
    for (int c = 0; c < C; ++c) s += fabsf(mx[c] - my[c]);
    v += s / (float)C;
  }
  loss[b] += v;
}

// ---------------- covariance abs-diff mean --------------------------------------
__global__ void k_covdiff(const float* __restrict__ cov, float* __restrict__ loss,
                          int C, int ns) {
  int b = blockIdx.y;
  long CC = (long)C * C;
  const float* cx = cov + (size_t)b * CC;
  const float* cy = cov + (size_t)8 * CC;
  float s = 0.f;
  for (long idx = (long)blockIdx.x * blockDim.x + threadIdx.x; idx < CC;
       idx += (long)gridDim.x * blockDim.x)
    s += fabsf(cx[idx] - cy[idx]);
  __shared__ float red[256];
  red[threadIdx.x] = s;
  __syncthreads();
  for (int st = 128; st > 0; st >>= 1) {
    if (threadIdx.x < st) red[threadIdx.x] += red[threadIdx.x + st];
    __syncthreads();
  }
  if (threadIdx.x == 0)
    atomicAdd(&loss[b], red[0] / ((float)(ns - 1) * (float)CC));
}

// ---------------- image feature (c=3): YUV transform ----------------------------
__global__ void k_yuv(const _Float16* __restrict__ xs, float* __restrict__ yuv, int ns) {
  int bi = blockIdx.y;
  int i  = blockIdx.x * blockDim.x + threadIdx.x;
  if (i >= ns) return;
  const _Float16* p = xs + ((size_t)bi * ns + i) * 3;
  float x0 = (float)p[0], x1 = (float)p[1], x2 = (float)p[2];
  float* o = yuv + ((size_t)bi * ns + i) * 3;
  o[0] = x0 * 0.57735f + x1 * (-0.57735f)  + x2 * (-0.57735f);
  o[1] = x0 * 0.57735f + x1 * 0.788675f    + x2 * (-0.211325f);
  o[2] = x0 * 0.57735f + x1 * (-0.211325f) + x2 * 0.788675f;
}

// ---------------- image feature REMD over (l2/3 + cos), c=3 ---------------------
__global__ void k_img_min(const float* __restrict__ yuv, float* __restrict__ acc,
                          int ns, int mode) {
  int b = blockIdx.y;
  int i = blockIdx.x * blockDim.x + threadIdx.x;
  if (i >= ns) return;
  const float* X = yuv + (size_t)b * ns * 3;
  const float* Y = yuv + (size_t)8 * ns * 3;
  const float* F = (mode == 0) ? X : Y;   // fixed sample array
  const float* O = (mode == 0) ? Y : X;   // loop array
  float a0 = F[(size_t)i * 3], a1 = F[(size_t)i * 3 + 1], a2 = F[(size_t)i * 3 + 2];
  float aa = a0 * a0 + a1 * a1 + a2 * a2;
  float ra = sqrtf(aa);
  float best = 1e30f;
  for (int m = 0; m < ns; ++m) {
    float b0 = O[(size_t)m * 3], b1 = O[(size_t)m * 3 + 1], b2 = O[(size_t)m * 3 + 2];
    float bb = b0 * b0 + b1 * b1 + b2 * b2;
    float dot = a0 * b0 + a1 * b1 + a2 * b2;
    float l2 = fminf(fmaxf(aa + bb - 2.f * dot, 1e-5f), 1e5f) * (1.0f / 3.0f);
    float cs = 1.0f - dot / (ra * sqrtf(bb) + 1e-10f);
    best = fminf(best, l2 + cs);
  }
  atomicAdd(&acc[b], best / (float)ns);
}

// ---------------- final: mean over batch ----------------------------------------
__global__ void k_final(const float* __restrict__ loss, float* __restrict__ out) {
  if (threadIdx.x == 0) {
    float s = 0.f;
    for (int b = 0; b < 8; ++b) s += loss[b];
    out[0] = s * 0.125f;
  }
}

// ================================================================================
static inline int cdiv(int a, int b) { return (a + b - 1) / b; }

extern "C" void kernel_launch(void* const* d_in, const int* in_sizes, int n_in,
                              void* d_out, int out_size, void* d_ws, size_t ws_size,
                              hipStream_t stream) {
  (void)in_sizes; (void)n_in; (void)out_size; (void)ws_size;
  const float* gen = (const float*)d_in[0];
  const float* tgt = (const float*)d_in[1];

  static const int CIN[13]  = {3,64,64,128,128,256,256,256,512,512,512,512,512};
  static const int COUT[13] = {64,64,128,128,256,256,256,512,512,512,512,512,512};
  static const int TAP[13]  = {1,1,1,1,1,1,1,0,0,1,0,0,1};
  static const int POOL[13] = {0,1,0,1,0,0,1,0,0,1,0,0,0};

  // ---- workspace bump allocator ----
  char* ws = (char*)d_ws;
  size_t off = 0;
  auto alloc = [&](size_t bytes) -> char* {
    char* p = ws + off;
    off += (bytes + 255) & ~(size_t)255;
    return p;
  };
  const size_t ACT = (size_t)9 * 64 * 224 * 224 * sizeof(_Float16);  // max stage
  size_t woff[13]; size_t wtot = 0;
  for (int l = 0; l < 13; ++l) { woff[l] = wtot; wtot += (size_t)CIN[l] * 9 * COUT[l]; }

  _Float16* actA  = (_Float16*)alloc(ACT);
  _Float16* actB  = (_Float16*)alloc(ACT);
  _Float16* wpack = (_Float16*)alloc(wtot * sizeof(_Float16));
  _Float16* xs    = (_Float16*)alloc((size_t)9 * 1024 * 512 * sizeof(_Float16));
  _Float16* xsT   = (_Float16*)alloc((size_t)9 * 1024 * 512 * sizeof(_Float16));
  float*    norms = (float*)alloc((size_t)9 * 1024 * sizeof(float));
  float*    mu    = (float*)alloc((size_t)9 * 512 * sizeof(float));
  float*    dots  = (float*)alloc((size_t)8 * 1024 * 1024 * sizeof(float));
  float*    cov   = (float*)alloc((size_t)9 * 512 * 512 * sizeof(float));
  float*    yuv   = (float*)alloc((size_t)9 * 1024 * 3 * sizeof(float));
  float*    accR  = (float*)alloc(16 * sizeof(float));  // accR[8] + accC[8]
  float*    accC  = accR + 8;
  float*    loss  = (float*)alloc(8 * sizeof(float));

  // ---- 1. prep: normalize 9 images -> fp16 ----
  const int HW0 = 224 * 224;
  k_prep<<<2048, 256, 0, stream>>>(gen, tgt, actA, HW0);
  hipMemsetAsync(loss, 0, 8 * sizeof(float), stream);

  // ---- tap pipeline (features i >= 1) ----
  auto tap = [&](const _Float16* act, int C, int HW) {
    int ns = HW < 1024 ? HW : 1024;
    k_subsample<<<dim3(cdiv(ns, 256), 9), 256, 0, stream>>>(act, C, HW, ns, xs, xsT, norms);
    k_mean<<<dim3(cdiv(C, 256), 9), 256, 0, stream>>>(xsT, mu, C, ns);
    // pairwise dot matrix via WMMA GEMM: dots[b] = xs[b] (ns x C) * ys (ns x C)^T
    k_gemm_nt<<<dim3(cdiv(ns, 128), cdiv(ns, 64), 8), 256, 0, stream>>>(
        xs, xs + (size_t)8 * ns * C, dots, ns, ns, C,
        (long)ns * C, 0L, (long)ns * ns);
    hipMemsetAsync(accR, 0, 16 * sizeof(float), stream);
    k_cos_min<<<dim3(cdiv(ns, 256), 8), 256, 0, stream>>>(dots, norms, accR, ns, 0);
    k_cos_min<<<dim3(cdiv(ns, 256), 8), 256, 0, stream>>>(dots, norms, accC, ns, 1);
    k_remd_mu<<<1, 32, 0, stream>>>(accR, accC, mu, loss, C);
    // moment: center then batched covariance via WMMA GEMM (9 batches incl. target)
    k_center<<<1024, 256, 0, stream>>>(xsT, mu, C, ns);
    k_gemm_nt<<<dim3(cdiv(C, 128), cdiv(C, 64), 9), 256, 0, stream>>>(
        xsT, xsT, cov, C, C, ns, (long)C * ns, (long)C * ns, (long)C * C);
    k_covdiff<<<dim3(8, 8), 256, 0, stream>>>(cov, loss, C, ns);
  };

  // ---- 2. feature 0 = normalized image (c=3), l2+cos REMD in YUV ----
  {
    int ns = 1024;
    k_subsample<<<dim3(cdiv(ns, 256), 9), 256, 0, stream>>>(actA, 3, HW0, ns, xs, xsT, norms);
    k_yuv<<<dim3(cdiv(ns, 256), 9), 256, 0, stream>>>(xs, yuv, ns);
    hipMemsetAsync(accR, 0, 16 * sizeof(float), stream);
    k_img_min<<<dim3(cdiv(ns, 256), 8), 256, 0, stream>>>(yuv, accR, ns, 0);
    k_img_min<<<dim3(cdiv(ns, 256), 8), 256, 0, stream>>>(yuv, accC, ns, 1);
    k_remd_mu<<<1, 32, 0, stream>>>(accR, accC, mu, loss, 0);  // C=0: remd only
  }

  // ---- 3. VGG conv chain with fused ReLU, taps and avg-pools ----
  _Float16* cur = actA;
  _Float16* nxt = actB;
  int H = 224;
  for (int l = 0; l < 13; ++l) {
    const float* Wl = (const float*)d_in[2 + 2 * l];
    const float* bl = (const float*)d_in[3 + 2 * l];
    long wn = (long)CIN[l] * 9 * COUT[l];
    k_packw<<<cdiv((int)wn, 256), 256, 0, stream>>>(Wl, wpack + woff[l], CIN[l], COUT[l]);
    int HW = H * H;
    k_conv3x3<<<dim3(cdiv(HW, 128), cdiv(COUT[l], 64), 9), 256, 0, stream>>>(
        cur, wpack + woff[l], bl, nxt, CIN[l], COUT[l], H, H);
    { _Float16* t = cur; cur = nxt; nxt = t; }
    if (TAP[l]) tap(cur, COUT[l], HW);
    if (POOL[l]) {
      k_pool<<<2048, 256, 0, stream>>>(cur, nxt, COUT[l], H, H);
      { _Float16* t = cur; cur = nxt; nxt = t; }
      H >>= 1;
    }
  }

  // ---- 4. final reduction ----
  k_final<<<1, 32, 0, stream>>>(loss, (float*)d_out);
}
